// GraphHeteroAtt_11218454577276
// MI455X (gfx1250) — matrix operand order repaired and checked by reading, not compiled, not run on hardware
//
#include <hip/hip_runtime.h>
#include <stdint.h>

typedef __attribute__((ext_vector_type(16))) _Float16 v16h;
typedef __attribute__((ext_vector_type(4)))  _Float16 v4h;
typedef __attribute__((ext_vector_type(8)))  float    v8f;

__device__ __forceinline__ float lrelu(float x) { return x >= 0.0f ? x : 0.01f * x; }

// ---------------- B-fragment LDS layout ----------------
// Fragment (c,j) covers W[32c..32c+31][16j..16j+15]; in contiguous k order lane L
// holds column 16j+(L&15), K-values 32c+16*(L>>4)+e, e=0..15 -> 16 contiguous
// f16 (32B) per lane => reads are 2 x ds_load_b128.
__device__ __forceinline__ const v16h* frag_ptr(const _Float16* base, int nj,
                                                int c, int j, int lane) {
  return (const v16h*)(base + (((c * nj + j) * 32 + lane) << 4));
}

// Cooperative f32 -> f16 weight staging into fragment layout.
// Quads of 4 consecutive k at fixed n: global reads coalesced across threads,
// LDS writes are aligned ds_store_b64.
__device__ __forceinline__ void stage_weight_f16(const float* __restrict__ w,
                                                 _Float16* __restrict__ dst,
                                                 int ldn, int tid, int nthr) {
  const int nj = ldn >> 4;
  const int total = 32 * ldn;                 // (128/4) k-quads * ldn columns
  for (int q = tid; q < total; q += nthr) {
    int kq = q / ldn;
    int nn = q - kq * ldn;
    int k  = kq << 2;
    int c  = k >> 5;
    int hl = (k >> 4) & 1;
    int e0 = k & 15;
    int j  = nn >> 4;
    int L  = (hl << 4) | (nn & 15);
    const float* src = w + (size_t)k * ldn + nn;
    v4h quad;
    quad[0] = (_Float16)src[0];
    quad[1] = (_Float16)src[ldn];
    quad[2] = (_Float16)src[2 * ldn];
    quad[3] = (_Float16)src[3 * ldn];
    *(v4h*)(dst + (((c * nj + j) * 32 + L) << 4) + e0) = quad;
  }
}

// A-fragment (16x32 f16) in contiguous k order (k0+16*half+e) from a row-major f32 row pointer that
// is already offset to (row m, col k0 + 16*(lane>>4)).
__device__ __forceinline__ v16h a_frag_from(const float* row) {
  v16h a;
#pragma unroll
  for (int v = 0; v < 8; ++v) {
    int kk = 2 * v;
    a[2 * v]     = (_Float16)row[kk];
    a[2 * v + 1] = (_Float16)row[kk + 1];
  }
  return a;
}

__global__ void __launch_bounds__(256)
hetero_att_kernel(const float* __restrict__ vt,
                  const float* __restrict__ Hm,
                  const float* __restrict__ Wgv,
                  const float* __restrict__ Wgh,
                  const float* __restrict__ Wuh,
                  float* __restrict__ out,
                  int n, int ntiles) {
  __shared__ _Float16 sWgv[8192];     // 16 KB, staged once per block
  __shared__ _Float16 sWgh[8192];     // 16 KB, restaged per head
  __shared__ _Float16 sWuh[16384];    // 32 KB, restaged per head
  __shared__ float    sH[8][2048];    // 64 KB, per-wave 16x128 f32 H tile
  __shared__ float    sred[8][16];    // row remap scratch

  const int tid  = threadIdx.x;
  const int lane = tid & 31;
  const int wave = tid >> 5;
  const int hh   = lane >> 4;

  stage_weight_f16(Wgv, sWgv, 64, tid, 256);
  __syncthreads();

  for (int base = blockIdx.x * 8; base < ntiles; base += gridDim.x * 8) {
    const int  tile     = base + wave;
    const bool valid    = tile < ntiles;
    const int  node0    = tile * 16;
    const bool fullTile = valid && (node0 + 16 <= n);
    float*     myH      = sH[wave];
    const unsigned myHlds = (unsigned)(uintptr_t)(void*)myH;   // LDS byte offset

    // async-copy first 2KB (rows 0..3) of head 0's tile into this wave's LDS slot
    if (fullTile) {
      const float* g0 = Hm + (size_t)node0 * 128;
#pragma unroll
      for (int i = 0; i < 4; ++i)
        asm volatile("global_load_async_to_lds_b128 %0, %1, off"
                     :: "v"(myHlds + (unsigned)(i * 32 + lane) * 16),
                        "v"(g0 + (size_t)(i * 32 + lane) * 4)
                     : "memory");
    }

    // ---------- g_key = f16(vt_tile) @ f16(Wgv), overlaps the async copy ----------
    v8f gkey[4];
    {
      const int mlim = valid ? (n - 1 - node0) : 0;
      const float* vbase = valid ? (vt + (size_t)node0 * 128) : vt;
      int m = lane & 15; if (m > mlim) m = mlim;
      v16h a_vt[4];
#pragma unroll
      for (int c = 0; c < 4; ++c)
        a_vt[c] = a_frag_from(vbase + (size_t)m * 128 + c * 32 + hh * 16);
#pragma unroll
      for (int j = 0; j < 4; ++j) {
        v8f acc = {};
#pragma unroll
        for (int c = 0; c < 4; ++c)
          acc = __builtin_amdgcn_wmma_f32_16x16x32_f16(
              false, a_vt[c], false, *frag_ptr(sWgv, 4, c, j, lane),
              (short)0, acc, false, false);
        gkey[j] = acc;
      }
    }

    v8f uacc[8];
#pragma unroll
    for (int j = 0; j < 8; ++j) { v8f z = {}; uacc[j] = z; }
    float wsum = 0.0f;

    for (int k = 0; k < 8; ++k) {
      // cooperative per-head weight restage (f32 L2 -> f16 LDS fragments)
      stage_weight_f16(Wgh + (size_t)k * 8192,  sWgh, 64,  tid, 256);
      stage_weight_f16(Wuh + (size_t)k * 16384, sWuh, 128, tid, 256);
      __syncthreads();

      // -------- A fragments for this head --------
      v16h a_h[4];
      if (fullTile) {
        asm volatile("s_wait_asynccnt 0x0" ::: "memory");
#pragma unroll
        for (int c = 0; c < 4; ++c)
          a_h[c] = a_frag_from(myH + (lane & 15) * 128 + c * 32 + hh * 16);
        asm volatile("s_wait_dscnt 0x0" ::: "memory");  // tile reads done
        if (k < 7) {                                    // prefetch next head
          const float* gn = Hm + ((size_t)(k + 1) * (size_t)n + (size_t)node0) * 128;
#pragma unroll
          for (int i = 0; i < 4; ++i)
            asm volatile("global_load_async_to_lds_b128 %0, %1, off"
                         :: "v"(myHlds + (unsigned)(i * 32 + lane) * 16),
                            "v"(gn + (size_t)(i * 32 + lane) * 4)
                         : "memory");
        }
      } else {  // tail / idle wave: clamped direct loads
        const int mlim = valid ? (n - 1 - node0) : 0;
        const float* hb = valid ? (Hm + ((size_t)k * (size_t)n + (size_t)node0) * 128) : Hm;
        int m = lane & 15; if (m > mlim) m = mlim;
#pragma unroll
        for (int c = 0; c < 4; ++c)
          a_h[c] = a_frag_from(hb + (size_t)m * 128 + c * 32 + hh * 16);
      }

      // -------- g_query + fused row-dot with g_key --------
      float s[8];
#pragma unroll
      for (int v = 0; v < 8; ++v) s[v] = 0.0f;
#pragma unroll
      for (int j = 0; j < 4; ++j) {
        v8f acc = {};
#pragma unroll
        for (int c = 0; c < 4; ++c)
          acc = __builtin_amdgcn_wmma_f32_16x16x32_f16(
              false, a_h[c], false, *frag_ptr(sWgh, 4, c, j, lane),
              (short)0, acc, false, false);
#pragma unroll
        for (int v = 0; v < 8; ++v) s[v] += acc[v] * gkey[j][v];
      }
#pragma unroll
      for (int v = 0; v < 8; ++v) {
        s[v] += __shfl_xor(s[v], 1, 16);
        s[v] += __shfl_xor(s[v], 2, 16);
        s[v] += __shfl_xor(s[v], 4, 16);
        s[v] += __shfl_xor(s[v], 8, 16);
      }
      // C-layout rows (v+8*half) -> A-layout rows (lane&15), wave-private LDS
#pragma unroll
      for (int v = 0; v < 8; ++v)
        if ((lane & 15) == v) sred[wave][v + 8 * hh] = s[v];
      float gval = sred[wave][lane & 15];

      float w = __expf(lrelu(gval) * 0.125f);     // 1/sqrt(G_DIM) = 1/8
      wsum += w;

      // fold unnormalized weight into A rows: (diag(w) * H_k) @ Wuh_k
      _Float16 wh = (_Float16)w;
#pragma unroll
      for (int c = 0; c < 4; ++c)
#pragma unroll
        for (int e = 0; e < 16; ++e) a_h[c][e] *= wh;

#pragma unroll
      for (int j = 0; j < 8; ++j)
#pragma unroll
        for (int c = 0; c < 4; ++c)
          uacc[j] = __builtin_amdgcn_wmma_f32_16x16x32_f16(
              false, a_h[c], false, *frag_ptr(sWuh, 8, c, j, lane),
              (short)0, uacc[j], false, false);

      __syncthreads();   // weights consumed; safe to restage next head
    }

    // ---------- normalize by head-sum, leaky-relu, store ----------
    sred[wave][lane & 15] = wsum;   // lanes L and L+16 write identical values
    float winv[8];
#pragma unroll
    for (int v = 0; v < 8; ++v) winv[v] = 1.0f / sred[wave][v + 8 * hh];

#pragma unroll
    for (int v = 0; v < 8; ++v) {
      int node = node0 + v + 8 * hh;
      if (valid && node < n) {
        float* orow = out + (size_t)node * 128 + (lane & 15);
#pragma unroll
        for (int j = 0; j < 8; ++j)
          orow[j * 16] = lrelu(uacc[j][v] * winv[v]);
      }
    }
  }
}

extern "C" void kernel_launch(void* const* d_in, const int* in_sizes, int n_in,
                              void* d_out, int out_size, void* d_ws, size_t ws_size,
                              hipStream_t stream) {
  const float* vt  = (const float*)d_in[0];
  const float* Hm  = (const float*)d_in[1];
  const float* Wgv = (const float*)d_in[2];
  const float* Wgh = (const float*)d_in[3];
  const float* Wuh = (const float*)d_in[4];
  float* out = (float*)d_out;

  int n = in_sizes[0] / 128;                // V_DIM = 128
  int ntiles = (n + 15) / 16;               // 6250 for N = 100000
  int blocks = (ntiles + 7) / 8;            // 8 wave32 tiles per block
  if (blocks > 16384) blocks = 16384;

  hetero_att_kernel<<<blocks, 256, 0, stream>>>(vt, Hm, Wgv, Wgh, Wuh, out,
                                                n, ntiles);
}
